// Attention_78280073937702
// MI455X (gfx1250) — compile-verified, matrix operand order checked by reading, not run on hardware
//
#include <hip/hip_runtime.h>

typedef unsigned short u16;
typedef __attribute__((ext_vector_type(16))) __bf16 v16bf;
typedef __attribute__((ext_vector_type(8)))  float  v8f;
typedef __attribute__((ext_vector_type(4)))  unsigned int v4u;
typedef __attribute__((ext_vector_type(4)))  int v4i_t;
typedef __attribute__((ext_vector_type(8)))  int v8i_t;

#define S_LEN  2048
#define HIDDEN 4096
#define NH     32
#define HD     128
#define QKV    (3 * HIDDEN)

union FragU { uint4 q[2]; v16bf v; };

__device__ __forceinline__ u16 f2bf(float f) {
  unsigned int u = __float_as_uint(f);
  u += 0x7FFFu + ((u >> 16) & 1u);   // round-to-nearest-even
  return (u16)(u >> 16);
}

__device__ __forceinline__ void cvt_store8(const float* __restrict__ g, u16* l) {
  union { u16 s[8]; uint4 q; } t;
#pragma unroll
  for (int i = 0; i < 8; ++i) t.s[i] = f2bf(g[i]);
  *(uint4*)l = t.q;
}

// 16x32 bf16 fragment, CDNA5 WMMA 16-bit A/B lane layout:
// lanes 0-15 : row = lane,    K = 0..7 and 16..23
// lanes 16-31: row = lane-16, K = 8..15 and 24..31
__device__ __forceinline__ v16bf frag_lds(const u16* tile, int ld) {  // row-major
  int lane = threadIdx.x & 31;
  const u16* p = tile + (lane & 15) * ld + (lane >> 4) * 8;
  FragU f;
  f.q[0] = *(const uint4*)p;
  f.q[1] = *(const uint4*)(p + 16);
  return f.v;
}

__device__ __forceinline__ v16bf frag_gbl(const u16* __restrict__ base, int ld) {
  int lane = threadIdx.x & 31;
  const u16* p = base + (size_t)(lane & 15) * ld + (lane >> 4) * 8;
  FragU f;
  f.q[0] = *(const uint4*)p;
  f.q[1] = *(const uint4*)(p + 16);
  return f.v;
}

__device__ __forceinline__ unsigned rfl_u(unsigned x) {
  return (unsigned)__builtin_amdgcn_readfirstlane((int)x);
}

// Tensor Data Mover: 2-D bf16 tile (tile_d0 contiguous elems x tile_d1 rows,
// row stride stride0 elems) from global -> LDS.  D# per CDNA5 ISA ch.8.
__device__ __forceinline__ void tdm_load_2d_bf16(unsigned lds_off, const u16* gptr,
                                                 unsigned tile_d0, unsigned tile_d1,
                                                 unsigned stride0) {
  unsigned long long ga = (unsigned long long)(uintptr_t)gptr;
  unsigned galo = rfl_u((unsigned)ga);
  unsigned gahi = rfl_u((unsigned)(ga >> 32));
  lds_off = rfl_u(lds_off);
  v4u g0;
  g0.x = 1u;                                   // count=1 (valid), user mode
  g0.y = lds_off;                              // lds_addr (bytes)
  g0.z = galo;                                 // global_addr[31:0]
  g0.w = (gahi & 0x01FFFFFFu) | 0x80000000u;   // global_addr[56:32] | type=2
  v8i_t g1;
  g1[0] = (int)(1u << 16);                     // data_size=1 -> 2 bytes
  g1[1] = (int)((tile_d0 & 0xFFFFu) << 16);                     // tensor_dim0[15:0]
  g1[2] = (int)((tile_d0 >> 16) | ((tile_d1 & 0xFFFFu) << 16)); // td0[31:16]|td1[15:0]
  g1[3] = (int)((tile_d1 >> 16) | (tile_d0 << 16));             // td1[31:16]|tile_dim0
  g1[4] = (int)(tile_d1 & 0xFFFFu);            // tile_dim1 (tile_dim2=0)
  g1[5] = (int)stride0;                        // tensor_dim0_stride[31:0]
  g1[6] = 0;                                   // stride0[47:32]=0, dim1_stride=0
  g1[7] = 0;
  v4i_t gz = {0, 0, 0, 0};
#if defined(__clang_major__) && (__clang_major__ >= 23)
  v8i_t gz8 = {0, 0, 0, 0, 0, 0, 0, 0};
  __builtin_amdgcn_tensor_load_to_lds(g0, g1, gz, gz, gz8, 0);
#else
  __builtin_amdgcn_tensor_load_to_lds(g0, g1, gz, gz, 0);
#endif
}

// C = A(MxK,f32,row-major) * B(NxK,f32,row-major)^T, f32 out.
// On-the-fly f32->bf16 conversion into LDS staging, bf16 WMMA, f32 acc.
__global__ __launch_bounds__(256)
void gemm_f32in_bf16wmma(const float* __restrict__ A, const float* __restrict__ B,
                         float* __restrict__ C, int M, int N, int K) {
  __shared__ __align__(16) u16 As[128 * 32];
  __shared__ __align__(16) u16 Bs[128 * 32];
  const int tid  = threadIdx.x;
  const int m0   = blockIdx.y * 128;
  const int n0   = blockIdx.x * 128;
  const int wave = tid >> 5;
  const int wm   = wave >> 2;
  const int wn   = wave & 3;
  const int lane = tid & 31, half = lane >> 4, col = lane & 15;

  v8f acc[4][2] = {};
  for (int kk = 0; kk < K; kk += 32) {
#pragma unroll
    for (int c = 0; c < 2; ++c) {           // 512 16B chunks / 256 threads
      int chunk = tid + c * 256;
      int row = chunk >> 2, cl = (chunk & 3) * 8;
      cvt_store8(A + (size_t)(m0 + row) * K + kk + cl, &As[row * 32 + cl]);
      cvt_store8(B + (size_t)(n0 + row) * K + kk + cl, &Bs[row * 32 + cl]);
    }
    if (kk + 32 < K) {                      // prefetch next K slab -> global_prefetch_b8
      int row0 = tid >> 2, cl0 = (tid & 3) * 8;
      __builtin_prefetch(A + (size_t)(m0 + row0) * K + kk + 32 + cl0, 0, 1);
      __builtin_prefetch(B + (size_t)(n0 + row0) * K + kk + 32 + cl0, 0, 1);
    }
    __syncthreads();
    v16bf af[4], bfr[2];
#pragma unroll
    for (int fi = 0; fi < 4; ++fi) af[fi]  = frag_lds(&As[(wm * 64 + fi * 16) * 32], 32);
#pragma unroll
    for (int fj = 0; fj < 2; ++fj) bfr[fj] = frag_lds(&Bs[(wn * 32 + fj * 16) * 32], 32);
#pragma unroll
    for (int fi = 0; fi < 4; ++fi)
#pragma unroll
      for (int fj = 0; fj < 2; ++fj)
        acc[fi][fj] = __builtin_amdgcn_wmma_f32_16x16x32_bf16(
            false, af[fi], false, bfr[fj], (short)0, acc[fi][fj], false, false);
    __syncthreads();
  }
#pragma unroll
  for (int fi = 0; fi < 4; ++fi)
#pragma unroll
    for (int fj = 0; fj < 2; ++fj)
#pragma unroll
      for (int i = 0; i < 8; ++i) {
        int gm = m0 + wm * 64 + fi * 16 + half * 8 + i;
        int gn = n0 + wn * 32 + fj * 16 + col;
        C[(size_t)gm * N + gn] = acc[fi][fj][i];
      }
}

// RoPE + layout scatter: proj f32 [s][12288] -> Q,K bf16 [h][s][d], V^T bf16 [h][d][s]
__global__ __launch_bounds__(128)
void rope_scatter(const float* __restrict__ proj, const int* __restrict__ pos,
                  u16* __restrict__ Qb, u16* __restrict__ Kb, u16* __restrict__ Vt) {
  const int head = blockIdx.x, s = blockIdx.y, d = threadIdx.x;
  const float* row = proj + (size_t)s * QKV + head * HD;
  float qv = row[d], kv = row[HIDDEN + d], vv = row[2 * HIDDEN + d];
  float qp = (d < 64) ? -row[d + 64] : row[d - 64];
  float kp = (d < 64) ? -row[HIDDEN + d + 64] : row[HIDDEN + d - 64];
  float p    = (float)pos[s];
  float e    = (float)(d & 63) * (1.0f / 64.0f);
  float ang  = p * __powf(10000.0f, -e);
  float co = cosf(ang), si = sinf(ang);
  size_t hb = (size_t)head * S_LEN * HD;
  Qb[hb + (size_t)s * HD + d]    = f2bf(qv * co + qp * si);
  Kb[hb + (size_t)s * HD + d]    = f2bf(kv * co + kp * si);
  Vt[hb + (size_t)d * S_LEN + s] = f2bf(vv);
}

// Flash attention: 1 wave = one 16-query tile, 32-key chunks, K/V staged via TDM.
__global__ __launch_bounds__(64)
void flash_attn(const u16* __restrict__ Qb, const u16* __restrict__ Kb,
                const u16* __restrict__ Vt, float* __restrict__ attn) {
  __shared__ __align__(16) u16 Klds[2][32 * HD];   // 2 x 8 KB
  __shared__ __align__(16) u16 Vlds[2][HD * 32];   // 2 x 8 KB
  __shared__ __align__(16) u16 Plds[2][16 * 32];   // 2 x 1 KB
  const int w    = threadIdx.x >> 5;
  const int lane = threadIdx.x & 31;
  const int half = lane >> 4, col = lane & 15;
  const int t    = blockIdx.x * 2 + w;
  const int head = t >> 7;
  const int qt   = t & 127;
  const int q0   = qt * 16;
  const size_t hb = (size_t)head * S_LEN * HD;

  const unsigned koff = (unsigned)(uintptr_t)(void*)Klds[w];  // LDS byte offset
  const unsigned voff = (unsigned)(uintptr_t)(void*)Vlds[w];

  v16bf qf[4];
#pragma unroll
  for (int f = 0; f < 4; ++f)
    qf[f] = frag_gbl(Qb + hb + (size_t)q0 * HD + f * 32, HD);

  v8f o[8] = {};
  float mrow[8], lrow[8];
#pragma unroll
  for (int i = 0; i < 8; ++i) { mrow[i] = -3.0e38f; lrow[i] = 0.0f; }

  const float norm = 0.08838834764831845f;  // 1/sqrt(128)
  u16* pl = Plds[w];
  const int nchunks = (q0 + 16 + 31) >> 5;  // causal: keys 0 .. q0+15

  for (int c = 0; c < nchunks; ++c) {
    const int j0 = c * 32;
    // TDM: K tile 32 rows x 128 elems (row stride 128); V^T tile 128 rows x 32 (stride S)
    tdm_load_2d_bf16(koff, Kb + hb + (size_t)j0 * HD, HD, 32, HD);
    tdm_load_2d_bf16(voff, Vt + hb + j0, 32, HD, S_LEN);
    __builtin_amdgcn_s_wait_tensorcnt(0);
    asm volatile("" ::: "memory");

    v8f sc[2];
#pragma unroll
    for (int t2 = 0; t2 < 2; ++t2) {
      v8f a = {};
#pragma unroll
      for (int f = 0; f < 4; ++f) {
        v16bf kf = frag_lds(&Klds[w][(t2 * 16) * HD + f * 32], HD);
        a = __builtin_amdgcn_wmma_f32_16x16x32_bf16(false, qf[f], false, kf,
                                                    (short)0, a, false, false);
      }
      sc[t2] = a;
    }
#pragma unroll
    for (int i = 0; i < 8; ++i) {   // row m = half*8+i, key = j0 + t2*16 + col
      const int qidx = q0 + half * 8 + i;
      float s0 = sc[0][i] * norm + ((j0 + col)      > qidx ? -1.0e9f : 0.0f);
      float s1 = sc[1][i] * norm + ((j0 + 16 + col) > qidx ? -1.0e9f : 0.0f);
      float cm = fmaxf(s0, s1);
#pragma unroll
      for (int off = 1; off < 16; off <<= 1)        // row reduce in 16-lane half
        cm = fmaxf(cm, __shfl_xor(cm, off, 32));
      float nm    = fmaxf(mrow[i], cm);
      float alpha = __expf(mrow[i] - nm);
      mrow[i] = nm;
      float p0 = __expf(s0 - nm), p1 = __expf(s1 - nm);
      float ps = p0 + p1;
#pragma unroll
      for (int off = 1; off < 16; off <<= 1)
        ps += __shfl_xor(ps, off, 32);
      lrow[i] = lrow[i] * alpha + ps;
#pragma unroll
      for (int f = 0; f < 8; ++f) o[f][i] *= alpha;
      pl[(half * 8 + i) * 32 + col]      = f2bf(p0);
      pl[(half * 8 + i) * 32 + 16 + col] = f2bf(p1);
    }
    __threadfence_block();          // same-wave LDS store->load ordering
    v16bf pf = frag_lds(pl, 32);    // probs as A-fragment (16 q x 32 keys)
#pragma unroll
    for (int fd = 0; fd < 8; ++fd) {
      v16bf vf = frag_lds(&Vlds[w][(fd * 16) * 32], 32);
      o[fd] = __builtin_amdgcn_wmma_f32_16x16x32_bf16(false, pf, false, vf,
                                                      (short)0, o[fd], false, false);
    }
  }
#pragma unroll
  for (int i = 0; i < 8; ++i) {
    float inv = 1.0f / lrow[i];
    const int gq = q0 + half * 8 + i;
#pragma unroll
    for (int fd = 0; fd < 8; ++fd)
      attn[(size_t)gq * HIDDEN + head * HD + fd * 16 + col] = o[fd][i] * inv;
  }
}

extern "C" void kernel_launch(void* const* d_in, const int* in_sizes, int n_in,
                              void* d_out, int out_size, void* d_ws, size_t ws_size,
                              hipStream_t stream) {
  const float* hidden = (const float*)d_in[0];
  // d_in[1] = attention_mask (causal, applied analytically)
  const int*   pos = (const int*)d_in[2];
  const float* Wp  = (const float*)d_in[3];
  const float* Wo  = (const float*)d_in[4];
  float* out = (float*)d_out;
  char*  ws  = (char*)d_ws;

  const size_t PROJ_BYTES  = (size_t)S_LEN * QKV * 4;     // 96 MB
  const size_t HEADS_BYTES = (size_t)S_LEN * HIDDEN * 2;  // 16 MB each
  float* proj = (float*)ws;
  u16*   Qb   = (u16*)(ws + PROJ_BYTES);
  u16*   Kb   = (u16*)(ws + PROJ_BYTES + HEADS_BYTES);
  u16*   Vt   = (u16*)(ws + PROJ_BYTES + 2 * HEADS_BYTES);
  float* attn = (float*)(ws + PROJ_BYTES + 3 * HEADS_BYTES);

  // 1) QKV projection: [2048,4096] x [12288,4096]^T -> [2048,12288]
  gemm_f32in_bf16wmma<<<dim3(QKV / 128, S_LEN / 128), 256, 0, stream>>>(
      hidden, Wp, proj, S_LEN, QKV, HIDDEN);
  // 2) RoPE + head scatter (Q,K bf16 [h][s][d]; V^T bf16 [h][d][s])
  rope_scatter<<<dim3(NH, S_LEN), HD, 0, stream>>>(proj, pos, Qb, Kb, Vt);
  // 3) Flash attention -> attn f32 [s][4096]
  flash_attn<<<dim3(NH * (S_LEN / 16) / 2), 64, 0, stream>>>(Qb, Kb, Vt, attn);
  // 4) Output projection: [2048,4096] x [4096,4096]^T -> d_out
  gemm_f32in_bf16wmma<<<dim3(HIDDEN / 128, S_LEN / 128), 256, 0, stream>>>(
      attn, Wo, out, S_LEN, HIDDEN, HIDDEN);
}
